// LSTMFC_395136991536
// MI455X (gfx1250) — compile-verified, matrix operand order checked by reading, not run on hardware
//
#include <hip/hip_runtime.h>
#include <cstdint>

// ---------------- problem constants (match reference) ----------------
#define BATCH 256
#define TSTEPS 1024
#define INDIM 64
#define HID 256
#define OUTDIM 64
#define DELAY 20
#define NBLK 16      // 16 independent blocks, each owns 16 batch rows
#define ROWS 16      // batch rows per block (one WMMA M-tile)

// ---------------- workspace layout (bytes) ----------------
// WG_OFF : packed gate weights  bf16  [kT(12)][ntile(64)][lane(32)][e(16)]
// WO_OFF : packed W_out         bf16  [kT(8)][nT(4)][lane(32)][e(16)]
// XBF_OFF: input as bf16, [T][B][IN] (optional, if workspace large enough)
static constexpr size_t WG_OFF  = 0;
static constexpr size_t WG_N    = (size_t)12 * 64 * 32 * 16;          // 393216
static constexpr size_t WO_OFF  = WG_OFF + WG_N * 2;                  // 786432
static constexpr size_t WO_N    = (size_t)8 * 4 * 32 * 16;            // 16384
static constexpr size_t XBF_OFF = WO_OFF + WO_N * 2;                  // 819200
static constexpr size_t XBF_N   = (size_t)TSTEPS * BATCH * INDIM;     // 16777216
static constexpr size_t WS_NEED_X = XBF_OFF + XBF_N * 2;              // ~34.4 MB

// ---------------- LDS strides (bank-rotation padding) ----------------
#define HLD 264   // h row stride (256 + 8 pad)
#define FLD 72    // fb row stride (64 + 8 pad)

// ---------------- vector types for WMMA ----------------
typedef __attribute__((ext_vector_type(16))) __bf16 bf16x16;
typedef __attribute__((ext_vector_type(8)))  __bf16 bf16x8;
typedef __attribute__((ext_vector_type(8)))  float  f32x8;

__device__ __forceinline__ f32x8 wmma_bf16(bf16x16 a, bf16x16 b, f32x8 c) {
  return __builtin_amdgcn_wmma_f32_16x16x32_bf16(false, a, false, b, (short)0, c,
                                                 false, false);
}

// A fragment (16x32 bf16, row-major source, ISA 16-bit A layout):
// lane<16: elems 0..7 = K0..7, 8..15 = K16..23 ; lane>=16: K8..15 / K24..31
// Separate LDS and global helpers so address-space inference never sees a
// mixed-provenance pointer.
__device__ __forceinline__ bf16x16 ldA_lds(const __bf16* base, int ld, int k0,
                                           int lane) {
  const __bf16* p = base + (lane & 15) * ld + k0 + ((lane & 16) ? 8 : 0);
  bf16x8 lo = *reinterpret_cast<const bf16x8*>(p);
  bf16x8 hi = *reinterpret_cast<const bf16x8*>(p + 16);
  bf16x16 a;
#pragma unroll
  for (int i = 0; i < 8; ++i) { a[i] = lo[i]; a[i + 8] = hi[i]; }
  return a;
}

__device__ __forceinline__ bf16x16 ldA_glob(const __bf16* base, int ld, int k0,
                                            int lane) {
  const __bf16* p = base + (size_t)(lane & 15) * ld + k0 + ((lane & 16) ? 8 : 0);
  bf16x8 lo = *reinterpret_cast<const bf16x8*>(p);
  bf16x8 hi = *reinterpret_cast<const bf16x8*>(p + 16);
  bf16x16 a;
#pragma unroll
  for (int i = 0; i < 8; ++i) { a[i] = lo[i]; a[i + 8] = hi[i]; }
  return a;
}

// fp32 source with on-the-fly bf16 convert (fallback x path)
__device__ __forceinline__ bf16x16 ldA_f32(const float* p) {
  bf16x16 a;
#pragma unroll
  for (int i = 0; i < 8; ++i) { a[i] = (__bf16)p[i]; a[i + 8] = (__bf16)p[i + 16]; }
  return a;
}

__device__ __forceinline__ float sigm(float x) { return 1.0f / (1.0f + __expf(-x)); }
__device__ __forceinline__ float tanh_fast(float x) {
  return 1.0f - 2.0f / (__expf(2.0f * x) + 1.0f);
}

// K-index packing shared by A and B fragments (16-bit, K=32 tiles)
__device__ __forceinline__ int frag_kk(int half, int e) {
  return (e < 8) ? ((half ? 8 : 0) + e) : ((half ? 24 : 16) + (e - 8));
}

// ================= pack kernel (runs first on stream each call) =================
__global__ void lstm_pack(const float* __restrict__ input,
                          const float* __restrict__ W_ih_x,
                          const float* __restrict__ W_ih_fb,
                          const float* __restrict__ W_hh,
                          const float* __restrict__ W_out,
                          unsigned char* __restrict__ ws, int do_x) {
  const size_t tid0 = (size_t)blockIdx.x * blockDim.x + threadIdx.x;
  const size_t stride = (size_t)gridDim.x * blockDim.x;
  __bf16* wg = (__bf16*)(ws + WG_OFF);
  __bf16* wo = (__bf16*)(ws + WO_OFF);
  __bf16* xb = (__bf16*)(ws + XBF_OFF);

  // gate weights: B[k,n], k: [x(64)|fb(64)|h(256)], n = gate*256 + col
  for (size_t i = tid0; i < WG_N; i += stride) {
    int e = (int)(i & 15);
    int lane = (int)((i >> 4) & 31);
    int q = (int)(i >> 9);
    int ntile = q & 63;
    int kT = q >> 6;
    int k = kT * 32 + frag_kk(lane >> 4, e);
    int n = ntile * 16 + (lane & 15);
    float v = (k < 64)    ? W_ih_x[n * 64 + k]
              : (k < 128) ? W_ih_fb[n * 64 + (k - 64)]
                          : W_hh[n * 256 + (k - 128)];
    wg[i] = (__bf16)v;
  }
  // W_out: B[k,n] = W_out[n,k], K=256, N=64
  for (size_t i = tid0; i < WO_N; i += stride) {
    int e = (int)(i & 15);
    int lane = (int)((i >> 4) & 31);
    int q = (int)(i >> 9);
    int nT = q & 3;
    int kT = q >> 2;
    int k = kT * 32 + frag_kk(lane >> 4, e);
    int n = nT * 16 + (lane & 15);
    wo[i] = (__bf16)W_out[n * 256 + k];
  }
  // input fp32 [B,T,IN] -> bf16 [T,B,IN] (per-step tile contiguous)
  if (do_x) {
    for (size_t i = tid0; i < XBF_N; i += stride) {
      int k = (int)(i & 63);
      int b = (int)((i >> 6) & 255);
      int t = (int)(i >> 14);
      xb[i] = (__bf16)input[(size_t)b * (TSTEPS * INDIM) + (size_t)t * INDIM + k];
    }
  }
}

// ================= persistent recurrent kernel (no inter-block sync) =============
// Wave w owns gate N-tiles {g*16 + cc*8 + w : g in 0..3, cc in 0..1} so the
// i/f/g/o accumulators for a given h-column live in the SAME lane/row of the
// same wave -> LSTM elementwise runs directly on WMMA accumulators (no LDS
// gate exchange). Only h goes to LDS (next step's A operand + out-GEMM input).
template <bool XBF>
__global__ __launch_bounds__(256) void lstm_main(const float* __restrict__ input,
                                                 const float* __restrict__ b_ih,
                                                 const float* __restrict__ b_hh,
                                                 const float* __restrict__ b_out,
                                                 float* __restrict__ out,
                                                 const unsigned char* __restrict__ ws) {
  // static shared -> guaranteed addrspace(3), accesses lower to ds_* ops
  __shared__ __bf16 hls[ROWS * HLD];           // h state (bf16), ~8.3 KB
  __shared__ __bf16 fbls[DELAY * ROWS * FLD];  // delay ring (bf16), ~45 KB

  const int tid = threadIdx.x;
  const int blk = blockIdx.x;                 // owns batch rows [16*blk, 16*blk+16)
  const int b0 = blk * ROWS;
  const int wv = tid >> 5;                    // 8 waves
  const int lane = tid & 31;
  const int nl = lane & 15;
  const int rb = (lane & 16) ? 8 : 0;

  // zero block-local recurrent state (LDS is fresh every launch)
  for (int i = tid; i < ROWS * HLD; i += 256) hls[i] = (__bf16)0.0f;
  for (int i = tid; i < DELAY * ROWS * FLD; i += 256) fbls[i] = (__bf16)0.0f;

  // hoisted per-lane gate biases for the two owned column groups
  // col(cc) = (cc*8 + wv)*16 + nl
  float bi2[2], bf2[2], bg2[2], bo2[2];
#pragma unroll
  for (int cc = 0; cc < 2; ++cc) {
    const int col = (cc * 8 + wv) * 16 + nl;
    bi2[cc] = b_ih[col] + b_hh[col];
    bf2[cc] = b_ih[256 + col] + b_hh[256 + col];
    bg2[cc] = b_ih[512 + col] + b_hh[512 + col];
    bo2[cc] = b_ih[768 + col] + b_hh[768 + col];
  }
  const float boutv = (wv < 4) ? b_out[wv * 16 + nl] : 0.0f;

  float creg[ROWS];  // persistent cell state: (cc,r) -> row rb+r, col(cc)
#pragma unroll
  for (int r = 0; r < ROWS; ++r) creg[r] = 0.0f;

  __syncthreads();

  int slot = 0;  // rotating delay-ring slot (avoids %DELAY magic-number SALU)
  for (int t = 0; t < TSTEPS; ++t) {
    // Launder a ZERO offset (not the pointer!) each step: keeps the GEP chain
    // rooted at the global kernarg `ws` (so loads stay global_load with SGPR
    // base + immediate offsets) while making addresses unknowable across
    // iterations -> no LICM hoist/spill of the 96 loop-invariant B fragments.
    size_t lz = 0;
    asm volatile("" : "+s"(lz));
    const __bf16* wg = (const __bf16*)(ws + WG_OFF) + lz;
    const __bf16* wo = (const __bf16*)(ws + WO_OFF) + lz;

    // ---- phase 1: gates GEMM  [x_t | fb_t | h_{t-1}] (16x384) @ W (384x1024) ----
    const __bf16* fbs = fbls + slot * (ROWS * FLD);
    f32x8 acc[8];  // acc[g*2+cc]
#pragma unroll
    for (int q = 0; q < 8; ++q) acc[q] = (f32x8){};

#pragma unroll
    for (int kT = 0; kT < 12; ++kT) {
      bf16x16 a;  // one A fragment per K-tile, reused across 8 B tiles
      if (kT < 2) {
        if (XBF) {
          a = ldA_glob((const __bf16*)(ws + XBF_OFF) +
                           (size_t)t * (BATCH * INDIM) + (size_t)b0 * INDIM,
                       INDIM, kT * 32, lane);
        } else {
          const float* p = input + (size_t)(b0 + nl) * (TSTEPS * INDIM) +
                           (size_t)t * INDIM + kT * 32 + ((lane & 16) ? 8 : 0);
          a = ldA_f32(p);
        }
      } else if (kT < 4) {
        a = ldA_lds(fbs, FLD, (kT - 2) * 32, lane);
      } else {
        a = ldA_lds(hls, HLD, (kT - 4) * 32, lane);
      }
#pragma unroll
      for (int g = 0; g < 4; ++g)
#pragma unroll
        for (int cc = 0; cc < 2; ++cc) {
          const int ntile = g * 16 + cc * 8 + wv;
          bf16x16 bfr = *reinterpret_cast<const bf16x16*>(
              wg + ((size_t)(kT * 64 + ntile) * 32 + lane) * 16);
          acc[g * 2 + cc] = wmma_bf16(a, bfr, acc[g * 2 + cc]);
        }
    }
    __syncthreads();  // all hls/fbls reads of step t complete before h update

    // ---- phase 2: LSTM cell directly on accumulators; h -> LDS (bf16) ----
#pragma unroll
    for (int cc = 0; cc < 2; ++cc) {
      const int col = (cc * 8 + wv) * 16 + nl;
#pragma unroll
      for (int r = 0; r < 8; ++r) {
        const float gi = acc[0 * 2 + cc][r] + bi2[cc];
        const float gf = acc[1 * 2 + cc][r] + bf2[cc];
        const float gg = acc[2 * 2 + cc][r] + bg2[cc];
        const float go = acc[3 * 2 + cc][r] + bo2[cc];
        const float c = sigm(gf) * creg[cc * 8 + r] + sigm(gi) * tanh_fast(gg);
        creg[cc * 8 + r] = c;
        hls[(rb + r) * HLD + col] = (__bf16)(sigm(go) * tanh_fast(c));
      }
    }
    if (XBF && t + 1 < TSTEPS && tid < ROWS)
      __builtin_prefetch((const __bf16*)(ws + XBF_OFF) +
                             (size_t)(t + 1) * (BATCH * INDIM) +
                             (size_t)(b0 + tid) * INDIM,
                         0, 3);
    __syncthreads();  // h_t visible to all waves

    // ---- phase 3: output GEMM  out_t = h_t (16x256) @ W_out^T (256x64) ----
    // written into the delay ring (read back at t+DELAY) and the final output
    if (wv < 4) {
      f32x8 oacc = {};
#pragma unroll
      for (int kT = 0; kT < 8; ++kT) {
        bf16x16 a = ldA_lds(hls, HLD, kT * 32, lane);
        bf16x16 bfr = *reinterpret_cast<const bf16x16*>(
            wo + ((size_t)(kT * 4 + wv) * 32 + lane) * 16);
        oacc = wmma_bf16(a, bfr, oacc);
      }
      const int col = wv * 16 + nl;
      __bf16* sl = fbls + slot * (ROWS * FLD);  // slot just consumed this step
#pragma unroll
      for (int r = 0; r < 8; ++r) {
        const int b = rb + r;
        const float v = oacc[r] + boutv;
        out[((size_t)(b0 + b) * TSTEPS + t) * OUTDIM + col] = v;
        sl[b * FLD + col] = (__bf16)v;
      }
    }
    // no extra sync: next phase-1 only reads h / other ring slots; next
    // iteration's first barrier orders these writes against their consumers
    slot = (slot + 1 == DELAY) ? 0 : slot + 1;
  }
}

// ================= launcher =================
extern "C" void kernel_launch(void* const* d_in, const int* in_sizes, int n_in,
                              void* d_out, int out_size, void* d_ws, size_t ws_size,
                              hipStream_t stream) {
  (void)in_sizes; (void)n_in; (void)out_size;
  const float* input   = (const float*)d_in[0];
  const float* W_ih_x  = (const float*)d_in[1];
  const float* W_ih_fb = (const float*)d_in[2];
  const float* W_hh    = (const float*)d_in[3];
  const float* b_ih    = (const float*)d_in[4];
  const float* b_hh    = (const float*)d_in[5];
  const float* W_out   = (const float*)d_in[6];
  const float* b_out   = (const float*)d_in[7];
  float* out = (float*)d_out;
  unsigned char* ws = (unsigned char*)d_ws;

  const int do_x = (ws_size >= WS_NEED_X) ? 1 : 0;
  lstm_pack<<<512, 256, 0, stream>>>(input, W_ih_x, W_ih_fb, W_hh, W_out, ws, do_x);
  if (do_x)
    lstm_main<true><<<NBLK, 256, 0, stream>>>(input, b_ih, b_hh, b_out, out, ws);
  else
    lstm_main<false><<<NBLK, 256, 0, stream>>>(input, b_ih, b_hh, b_out, out, ws);
}
